// HumanDesignGNN_73074573574434
// MI455X (gfx1250) — compile-verified, matrix operand order checked by reading, not run on hardware
//
#include <hip/hip_runtime.h>
#include <hip/hip_bf16.h>
#include <math.h>

// ---------------- problem constants ----------------
#define N_NODES 64
#define HID     64
#define IN_DIM  34
#define NF_PER  (N_NODES * IN_DIM)       // 2176
#define N_LAYERS 3
#define N_HEADS  5
#define N_EDGES  128
#define OUT_PER  69
#define STR     65                       // f32 LDS row stride (bank-conflict-free)

#define USE_TDM 1                        // tensor_load_to_lds staging of node features

// ---------------- ws layout (in _Float16 elements) ----------------
#define WS_WP     0
#define WS_WSELF  4096
#define WS_WNEIGH 16384
#define WS_ATTN1  28672
#define WS_TOTAL  38912

typedef __attribute__((ext_vector_type(16))) _Float16 v16h;
typedef __attribute__((ext_vector_type(8)))  float    v8f;
typedef unsigned int u32;
typedef __attribute__((ext_vector_type(4))) u32 v4u;
typedef __attribute__((ext_vector_type(8))) u32 v8u;

// =====================================================================
// Prep: shuffle f32 weights into per-lane WMMA B-fragment order (f16).
// B-frag element e for lane (h=lane>>4, n=lane&15), tile (kt,nt):
//   K = kt*32 + (e&7) + 8*h + 16*(e>>3),  N = nt*16 + n
// =====================================================================
__global__ void prep_weights(const float* __restrict__ Wp,
                             const float* __restrict__ Wself,
                             const float* __restrict__ Wneigh,
                             const float* __restrict__ attn1W,
                             _Float16* __restrict__ ws) {
  int idx = blockIdx.x * blockDim.x + threadIdx.x;
  if (idx >= WS_TOTAL) return;
  float v = 0.0f;
  if (idx < WS_WSELF) {
    int r = idx;
    int e = r & 15, lane = (r >> 4) & 31, nk = r >> 9;
    int nt = nk & 3, kt = nk >> 2;
    int h = lane >> 4, n = lane & 15;
    int k = kt * 32 + (e & 7) + 8 * h + ((e >> 3) << 4);
    v = (k < IN_DIM) ? Wp[k * HID + nt * 16 + n] : 0.0f;
  } else if (idx < WS_WNEIGH) {
    int r = idx - WS_WSELF;
    int l = r >> 12; r &= 4095;
    int e = r & 15, lane = (r >> 4) & 31, nk = r >> 9;
    int nt = nk & 3, kt = nk >> 2;
    int h = lane >> 4, n = lane & 15;
    int k = kt * 32 + (e & 7) + 8 * h + ((e >> 3) << 4);
    v = Wself[l * 4096 + k * HID + nt * 16 + n];
  } else if (idx < WS_ATTN1) {
    int r = idx - WS_WNEIGH;
    int l = r >> 12; r &= 4095;
    int e = r & 15, lane = (r >> 4) & 31, nk = r >> 9;
    int nt = nk & 3, kt = nk >> 2;
    int h = lane >> 4, n = lane & 15;
    int k = kt * 32 + (e & 7) + 8 * h + ((e >> 3) << 4);
    v = Wneigh[l * 4096 + k * HID + nt * 16 + n];
  } else {
    int r = idx - WS_ATTN1;
    int hd = r >> 11; r &= 2047;
    int e = r & 15, lane = (r >> 4) & 31, nk = r >> 9;
    int nt = nk & 1, kt = nk >> 1;
    int h = lane >> 4, n = lane & 15;
    int k = kt * 32 + (e & 7) + 8 * h + ((e >> 3) << 4);
    v = attn1W[hd * 2048 + k * 32 + nt * 16 + n];
  }
  ws[idx] = (_Float16)v;
}

// compile-time column permutation inside a 32-wide K-half
#define FRAG_COL(kt, h, e) ((kt) * 32 + ((e) & 7) + 8 * (h) + (((e) >> 3) << 4))

// D = A1*B1 (+ A2*B2) + bias; A fragments are pre-packed f16 in LDS.
__device__ __forceinline__ void gemm_frag(const _Float16* A1, const _Float16* B1,
                                          const _Float16* A2, const _Float16* B2,
                                          const float* bias, float* D,
                                          int ntiles, int lane, int wave,
                                          const float* maskRow, bool do_relu) {
  int h = lane >> 4, n = lane & 15;
  int mbase = wave * 16;
  v16h a1[2], a2[2];
#pragma unroll
  for (int kt = 0; kt < 2; ++kt)
    a1[kt] = *(const v16h*)(A1 + (size_t)((wave * 2 + kt) * 32 + lane) * 16);
  if (maskRow) {
    _Float16 sc = (_Float16)maskRow[mbase + (lane & 15)];
#pragma unroll
    for (int kt = 0; kt < 2; ++kt)
#pragma unroll
      for (int e = 0; e < 16; ++e) a1[kt][e] *= sc;
  }
  if (A2) {
#pragma unroll
    for (int kt = 0; kt < 2; ++kt)
      a2[kt] = *(const v16h*)(A2 + (size_t)((wave * 2 + kt) * 32 + lane) * 16);
  }
  for (int nt = 0; nt < ntiles; ++nt) {
    float bv = bias ? bias[nt * 16 + n] : 0.0f;
    v8f c;
#pragma unroll
    for (int r = 0; r < 8; ++r) c[r] = bv;
#pragma unroll
    for (int kt = 0; kt < 2; ++kt) {
      v16h bf = *(const v16h*)(B1 + (size_t)((kt * ntiles + nt) * 32 + lane) * 16);
      c = __builtin_amdgcn_wmma_f32_16x16x32_f16(false, a1[kt], false, bf,
                                                 (short)0, c, false, false);
    }
    if (A2) {
#pragma unroll
      for (int kt = 0; kt < 2; ++kt) {
        v16h bf = *(const v16h*)(B2 + (size_t)((kt * ntiles + nt) * 32 + lane) * 16);
        c = __builtin_amdgcn_wmma_f32_16x16x32_f16(false, a2[kt], false, bf,
                                                   (short)0, c, false, false);
      }
    }
#pragma unroll
    for (int r = 0; r < 8; ++r) {
      float v = c[r];
      if (do_relu) v = fmaxf(v, 0.0f);
      D[(mbase + r + 8 * h) * STR + nt * 16 + n] = v;
    }
  }
}

__device__ __forceinline__ float sigmoidf(float x) { return 1.0f / (1.0f + expf(-x)); }

// =====================================================================
// Main kernel: one block (128 threads = 4 wave32) per batch element.
// =====================================================================
__global__ __launch_bounds__(128) void gnn_kernel(
    const float* __restrict__ nf, const float* __restrict__ sun,
    const int* __restrict__ eidx, const int* __restrict__ masks,
    const float* __restrict__ bp,
    const float* __restrict__ bconv, const float* __restrict__ lng,
    const float* __restrict__ lnb, const float* __restrict__ Wcodon,
    const float* __restrict__ bcodon,
    const float* __restrict__ attn1b, const float* __restrict__ attn2W,
    const float* __restrict__ attn2b, const float* __restrict__ outW,
    const float* __restrict__ outb,
    const float* __restrict__ fW1, const float* __restrict__ fb1,
    const float* __restrict__ fW2, const float* __restrict__ fb2,
    const _Float16* __restrict__ ws, float* __restrict__ out) {
  int b    = blockIdx.x;
  int tid  = threadIdx.x;
  int lane = tid & 31;
  int wave = tid >> 5;
  // pack-pass coordinates: each thread owns (row=node, K-half=kt)
  int pnode = tid >> 1, pkt = tid & 1;
  int pg = pnode >> 4, pm = pnode & 15;
  size_t pbase = (size_t)((pg * 2 + pkt) * 32 + pm) * 16;   // +h*256 for h=1

  __shared__ float xs[N_NODES * STR];                          // x (f32)
  __shared__ float ts[N_NODES * STR];                          // nf staging / neigh f32
  __shared__ float hs[N_NODES * STR];                          // GEMM outputs
  __shared__ _Float16 xa[4096] __attribute__((aligned(32)));   // x, A-frag order f16
  __shared__ _Float16 ta[4096] __attribute__((aligned(32)));   // input/neigh, A-frag order
  __shared__ float degL[N_NODES], sv[N_NODES], wv[N_NODES], red[N_NODES];
  __shared__ float psum[128], pqsum[128];
  __shared__ float maskL[N_HEADS * N_NODES];
  __shared__ int   erow[N_EDGES], ecol[N_EDGES];
  __shared__ float sunL[72];
  __shared__ float scoreL[8];
  __shared__ float f1a[2][32];

#if USE_TDM
  // ---- issue TDM: node_features[b] (2176 f32, contiguous) -> ts ----
  if (tid < 32) {
    unsigned lds_off = (unsigned)(size_t)(void*)ts;
    unsigned long long ga = (unsigned long long)(const void*)(nf + (size_t)b * NF_PER);
    v4u g0;
    g0[0] = 1u;                                        // count=1 (valid), user mode
    g0[1] = lds_off;                                   // lds_addr (bytes)
    g0[2] = (u32)(ga & 0xFFFFFFFFu);                   // global_addr[31:0]
    g0[3] = (u32)((ga >> 32) & 0x1FFFFFFu) | 0x80000000u; // addr[56:32] | type=2
    v8u g1;
    g1[0] = 0x20000u;                                  // data_size=2 (4B), mask=0
    g1[1] = (u32)NF_PER << 16;                         // tensor_dim0[15:0]
    g1[2] = 1u << 16;                                  // tensor_dim1 = 1
    g1[3] = (u32)NF_PER << 16;                         // tile_dim0 = 2176
    g1[4] = 1u;                                        // tile_dim1 = 1
    g1[5] = (u32)NF_PER;                               // tensor_dim0_stride
    g1[6] = 0u;
    g1[7] = 0u;
    asm volatile("tensor_load_to_lds %0, %1" :: "s"(g0), "s"(g1) : "memory");
  }
#else
  {
    const float* nfb = nf + (size_t)b * NF_PER;
    for (int i = tid; i < NF_PER; i += 128) ts[i] = nfb[i];
  }
#endif

  // ---- block-wide setup (overlaps with TDM) ----
  erow[tid] = eidx[tid];
  ecol[tid] = eidx[N_EDGES + tid];
  for (int i = tid; i < N_HEADS * N_NODES; i += 128) maskL[i] = (float)masks[i];
  for (int i = tid; i < 70; i += 128) sunL[i] = sun[(size_t)b * 70 + i];
  if (tid < N_NODES) degL[tid] = 0.0f;
  __syncthreads();
  atomicAdd(&degL[ecol[tid]], 1.0f);
  __syncthreads();
  if (tid < N_NODES) degL[tid] = 1.0f / fmaxf(degL[tid], 1.0f);

#if USE_TDM
  if (tid < 32) __builtin_amdgcn_s_wait_tensorcnt(0);
#endif
  __syncthreads();

  // ---- pack projection input fragments (zero-pad K 34->64); 2 b128-stores/h ----
  {
    const float* rowf = ts + pnode * IN_DIM;
#pragma unroll
    for (int h = 0; h < 2; ++h) {
      v16h v;
#pragma unroll
      for (int e = 0; e < 16; ++e) {
        int col = FRAG_COL(pkt, h, e);
        v[e] = (col < IN_DIM) ? (_Float16)rowf[col] : (_Float16)0.0f;
      }
      *(v16h*)(ta + pbase + h * 256) = v;
    }
  }
  __syncthreads();

  // ---- projection: x = relu(nf @ Wp + bp) ----
  gemm_frag(ta, ws + WS_WP, nullptr, nullptr, bp, xs, 4, lane, wave, nullptr, true);
  __syncthreads();
  {   // pack x into A-fragment f16 layout
    const float* rowf = xs + pnode * STR;
#pragma unroll
    for (int h = 0; h < 2; ++h) {
      v16h v;
#pragma unroll
      for (int e = 0; e < 16; ++e) v[e] = (_Float16)rowf[FRAG_COL(pkt, h, e)];
      *(v16h*)(xa + pbase + h * 256) = v;
    }
  }
  __syncthreads();

  // ---- GNN layers ----
  for (int l = 0; l < N_LAYERS; ++l) {
    __builtin_prefetch(ws + WS_WSELF + l * 4096, 0, 1);
    __builtin_prefetch(ws + WS_WNEIGH + l * 4096, 0, 1);
    for (int i = tid; i < N_NODES * HID; i += 128) ts[(i >> 6) * STR + (i & 63)] = 0.0f;
    __syncthreads();
    {   // scatter-add over edges: ts[col] += xs[row]
      int rs = erow[tid], cd = ecol[tid];
      for (int j = 0; j < HID; ++j)
        atomicAdd(&ts[cd * STR + j], xs[rs * STR + j]);
    }
    __syncthreads();
    {   // scale by 1/deg, pack neigh fragments
      float rd = degL[pnode];
      const float* rowf = ts + pnode * STR;
#pragma unroll
      for (int h = 0; h < 2; ++h) {
        v16h v;
#pragma unroll
        for (int e = 0; e < 16; ++e) v[e] = (_Float16)(rowf[FRAG_COL(pkt, h, e)] * rd);
        *(v16h*)(ta + pbase + h * 256) = v;
      }
    }
    __syncthreads();
    // h = x@Wself + neigh@Wneigh + bconv
    gemm_frag(xa, ws + WS_WSELF + l * 4096, ta, ws + WS_WNEIGH + l * 4096,
              bconv + l * HID, hs, 4, lane, wave, nullptr, false);
    __syncthreads();
    {   // LayerNorm partial stats: 2 threads per node
      const float* hr = hs + pnode * STR + pkt * 32;
      float s = 0.0f, q = 0.0f;
      for (int j = 0; j < 32; ++j) { float v = hr[j]; s += v; q += v * v; }
      psum[tid] = s; pqsum[tid] = q;
    }
    __syncthreads();
    {   // apply LN + residual relu; update xs and pack xa in one sweep
      float s = psum[pnode * 2] + psum[pnode * 2 + 1];
      float q = pqsum[pnode * 2] + pqsum[pnode * 2 + 1];
      float mu = s * (1.0f / HID);
      float var = q * (1.0f / HID) - mu * mu;
      float rsv = rsqrtf(var + 1e-5f);
      const float* hr = hs + pnode * STR;
      float* xr = xs + pnode * STR;
#pragma unroll
      for (int h = 0; h < 2; ++h) {
        v16h v;
#pragma unroll
        for (int e = 0; e < 16; ++e) {
          int col = FRAG_COL(pkt, h, e);
          float t = (hr[col] - mu) * rsv * lng[l * HID + col] + lnb[l * HID + col];
          float nx = xr[col] + fmaxf(t, 0.0f);
          xr[col] = nx;
          v[e] = (_Float16)nx;
        }
        *(v16h*)(xa + pbase + h * 256) = v;
      }
    }
    __syncthreads();
  }

  // ---- codons = sigmoid(x @ Wcodon + b) ----
  if (tid < N_NODES) {
    float s = bcodon[0];
    const float* xr = xs + tid * STR;
    for (int j = 0; j < HID; ++j) s += xr[j] * Wcodon[j];
    out[(size_t)b * OUT_PER + tid] = sigmoidf(s);
  }

  // ---- awareness heads ----
  for (int hd = 0; hd < N_HEADS; ++hd) {
    gemm_frag(xa, ws + WS_ATTN1 + hd * 2048, nullptr, nullptr,
              attn1b + hd * 32, hs, 2, lane, wave,
              maskL + hd * N_NODES, false);
    __syncthreads();
    {   // tanh-dot with attn2_W: 2 threads per node
      const float* hr = hs + pnode * STR + pkt * 16;
      const float* w2 = attn2W + hd * 32 + pkt * 16;
      float s = 0.0f;
      for (int k = 0; k < 16; ++k) s += tanhf(hr[k]) * w2[k];
      psum[tid] = s;
    }
    __syncthreads();
    if (tid < N_NODES)
      sv[tid] = psum[tid * 2] + psum[tid * 2 + 1] + attn2b[hd] +
                (1.0f - maskL[hd * N_NODES + tid]) * (-1.0e9f);
    __syncthreads();
    if (tid < N_NODES) {   // softmax over 64 nodes
      float mx = sv[0];
      for (int n2 = 1; n2 < N_NODES; ++n2) mx = fmaxf(mx, sv[n2]);
      float sum = 0.0f;
      for (int n2 = 0; n2 < N_NODES; ++n2) sum += expf(sv[n2] - mx);
      wv[tid] = expf(sv[tid] - mx) / sum;
    }
    __syncthreads();
    {   // pooled[j] = sum_n w_n * mask_n * x[n][j]: 2 threads per column
      float p = 0.0f;
      for (int n2 = pkt * 32; n2 < pkt * 32 + 32; ++n2)
        p += wv[n2] * maskL[hd * N_NODES + n2] * xs[n2 * STR + pnode];
      pqsum[tid] = p;
    }
    __syncthreads();
    if (tid < N_NODES)
      red[tid] = (pqsum[tid * 2] + pqsum[tid * 2 + 1]) * outW[hd * HID + tid];
    __syncthreads();
    if (tid == 0) {
      float s = outb[hd];
      for (int j = 0; j < HID; ++j) s += red[j];
      scoreL[hd] = sigmoidf(s);
    }
    __syncthreads();
  }

  // ---- FiLM heads on sun_encoding ----
  if (tid < 64) {
    int idx = tid >> 5, k = tid & 31;
    float a = fb1[idx * 32 + k];
    const float* w1 = fW1 + (size_t)idx * 70 * 32;
    for (int d = 0; d < 70; ++d) a += sunL[d] * w1[d * 32 + k];
    f1a[idx][k] = fmaxf(a, 0.0f);
  }
  __syncthreads();
  if (tid < 3) out[(size_t)b * OUT_PER + 64 + tid] = scoreL[tid];
  if (tid < 2) {
    int idx = tid;
    float g = fb2[idx * 2 + 0], bt = fb2[idx * 2 + 1];
    const float* w2 = fW2 + idx * 64;
    for (int k = 0; k < 32; ++k) {
      g  += f1a[idx][k] * w2[k * 2 + 0];
      bt += f1a[idx][k] * w2[k * 2 + 1];
    }
    out[(size_t)b * OUT_PER + 67 + idx] = sigmoidf(g * scoreL[3 + idx] + bt);
  }
}

// =====================================================================
extern "C" void kernel_launch(void* const* d_in, const int* in_sizes, int n_in,
                              void* d_out, int out_size, void* d_ws, size_t ws_size,
                              hipStream_t stream) {
  (void)n_in; (void)out_size; (void)ws_size;
  const float* nf      = (const float*)d_in[0];
  const float* sun     = (const float*)d_in[1];
  const int*   eidx    = (const int*)d_in[2];
  const int*   masks   = (const int*)d_in[3];
  const float* Wp      = (const float*)d_in[4];
  const float* bp      = (const float*)d_in[5];
  const float* Wself   = (const float*)d_in[6];
  const float* Wneigh  = (const float*)d_in[7];
  const float* bconv   = (const float*)d_in[8];
  const float* lng     = (const float*)d_in[9];
  const float* lnb     = (const float*)d_in[10];
  const float* Wcodon  = (const float*)d_in[11];
  const float* bcodon  = (const float*)d_in[12];
  const float* attn1W  = (const float*)d_in[13];
  const float* attn1b  = (const float*)d_in[14];
  const float* attn2W  = (const float*)d_in[15];
  const float* attn2b  = (const float*)d_in[16];
  const float* outW    = (const float*)d_in[17];
  const float* outb    = (const float*)d_in[18];
  const float* fW1     = (const float*)d_in[19];
  const float* fb1     = (const float*)d_in[20];
  const float* fW2     = (const float*)d_in[21];
  const float* fb2     = (const float*)d_in[22];
  float* out = (float*)d_out;
  _Float16* ws = (_Float16*)d_ws;

  int Bn = in_sizes[0] / NF_PER;

  prep_weights<<<(WS_TOTAL + 255) / 256, 256, 0, stream>>>(Wp, Wself, Wneigh, attn1W, ws);
  gnn_kernel<<<Bn, 128, 0, stream>>>(nf, sun, eidx, masks, bp, bconv, lng, lnb,
                                     Wcodon, bcodon, attn1b, attn2W, attn2b,
                                     outW, outb, fW1, fb1, fW2, fb2, ws, out);
}
